// RoPEAttention_52175262712500
// MI455X (gfx1250) — compile-verified, matrix operand order checked by reading, not run on hardware
//
#include <hip/hip_runtime.h>

#define DI __device__ __forceinline__

typedef _Float16 h8  __attribute__((ext_vector_type(8)));
typedef _Float16 h16 __attribute__((ext_vector_type(16)));
typedef float    f8  __attribute__((ext_vector_type(8)));

constexpr int B_  = 2;
constexpr int T_  = 2048;
constexpr int H_  = 1024;
constexpr int NH_ = 16;
constexpr int HD_ = 64;
constexpr int ROWS = B_ * T_;     // 4096
constexpr int QKVN = 3 * H_;      // 3072

// workspace offsets (in _Float16 elements); total = 48 MiB
constexpr size_t OFF_XH   = 0;
constexpr size_t OFF_WQKV = OFF_XH   + (size_t)ROWS * H_;
constexpr size_t OFF_WPRJ = OFF_WQKV + (size_t)QKVN * H_;
constexpr size_t OFF_Q    = OFF_WPRJ + (size_t)H_ * H_;
constexpr size_t OFF_K    = OFF_Q    + (size_t)B_ * NH_ * T_ * HD_;
constexpr size_t OFF_VT   = OFF_K    + (size_t)B_ * NH_ * T_ * HD_;
constexpr size_t OFF_ATT  = OFF_VT   + (size_t)B_ * NH_ * T_ * HD_;

// ---------------------------------------------------------------------------
// A-fragment loader for V_WMMA_F32_16X16X32_F16.
// ISA 16-bit A-matrix 16x32 layout: lane L holds row (L&15);
//   lanes 0-15 : K = {0..7, 16..23},  lanes 16-31 : K = {8..15, 24..31}
// => two contiguous 16-byte loads per lane from a row-major [16 x K] tile.
// B fragments (K x 16, column n = lane&15) use the identical pattern when the
// source is stored n-major over K — hence the pre-transposed weights and the
// [head, d, t] V layout.
// ---------------------------------------------------------------------------
DI h16 load_frag(const _Float16* base, int ld, int k0, int lane) {
  const _Float16* p = base + (size_t)(lane & 15) * ld + k0 + ((lane >> 4) << 3);
  h8 lo = *(const h8*)(p);        // K = base .. base+7
  h8 hi = *(const h8*)(p + 16);   // K = base+16 .. base+23
  return __builtin_shufflevector(lo, hi, 0, 1, 2, 3, 4, 5, 6, 7,
                                 8, 9, 10, 11, 12, 13, 14, 15);
}

DI f8 wmma_f16(h16 a, h16 b, f8 c) {
  return __builtin_amdgcn_wmma_f32_16x16x32_f16(false, a, false, b,
                                                (short)0, c, false, false);
}

// ---------------------------------------------------------------------------
// Precision staging: fp32 -> f16 cast, and cast+transpose for weights
// ---------------------------------------------------------------------------
__global__ void k_cast(const float* __restrict__ in, _Float16* __restrict__ out, int n) {
  int i = blockIdx.x * blockDim.x + threadIdx.x;
  if (i < n) out[i] = (_Float16)in[i];
}

// in: [K x N] fp32 row-major  ->  out: [N x K] f16 row-major
__global__ void k_transpose(const float* __restrict__ in, _Float16* __restrict__ out,
                            int K, int N) {
  int i = blockIdx.x * blockDim.x + threadIdx.x;
  if (i < K * N) {
    int n = i / K;
    int k = i - n * K;
    out[i] = (_Float16)in[(size_t)k * N + n];
  }
}

// ---------------------------------------------------------------------------
// RoPE epilogue for one 16x16 C tile. C-layout: lane holds
// (M = r + 8*(lane>>4), N = lane&15) in VGPR r; even/odd pair is in lane^1.
// which/head are wave-uniform per tile, d is per lane (tiles never straddle boundaries).
// ---------------------------------------------------------------------------
DI void rope_store(f8 acc, int rowbase, int lane, int which, int head, int d,
                   const float* __restrict__ rcos, const float* __restrict__ rsin,
                   _Float16* __restrict__ qb, _Float16* __restrict__ kb,
                   _Float16* __restrict__ vtb) {
  int hi = lane >> 4;
#pragma unroll
  for (int r = 0; r < 8; ++r) {
    int row = rowbase + r + 8 * hi;     // global token row in [0, 4096)
    int b_  = row >> 11;
    int t   = row & (T_ - 1);
    int hb  = b_ * NH_ + head;
    float v = acc[r];
    float partner = __shfl_xor(v, 1, 32);   // paired even/odd dim value
    if (which == 2) {
      vtb[((size_t)hb * HD_ + d) * T_ + t] = (_Float16)v;
    } else {
      float c = rcos[t * (HD_ / 2) + (d >> 1)];
      float s = rsin[t * (HD_ / 2) + (d >> 1)];
      float o = (d & 1) ? (partner * s + v * c)   // odd:  e*s + o*c
                        : (v * c - partner * s);  // even: e*c - o*s
      if (which == 0) o *= 0.125f;                // HD^-0.5 pre-scale on Q
      _Float16* dst = (which == 0) ? qb : kb;
      dst[((size_t)hb * T_ + t) * HD_ + d] = (_Float16)o;
    }
  }
}

// ---------------------------------------------------------------------------
// QKV GEMM (4096x3072, K=1024) with fused RoPE.
// 32x32 register block per wave: 2 A-frags x 2 B-frags -> 4 WMMAs per K-step,
// halving fragment loads per WMMA vs a 16x16 tile.
// ---------------------------------------------------------------------------
__global__ void k_qkv_rope(const _Float16* __restrict__ xh,
                           const _Float16* __restrict__ wT,   // [3072 x 1024]
                           const float* __restrict__ rcos,
                           const float* __restrict__ rsin,
                           _Float16* __restrict__ qb,         // [B*NH, T, HD]
                           _Float16* __restrict__ kb,         // [B*NH, T, HD]
                           _Float16* __restrict__ vtb) {      // [B*NH, HD, T]
  int lane = threadIdx.x & 31;
  int wid  = blockIdx.x * (blockDim.x >> 5) + (threadIdx.x >> 5);
  int mt = wid / (QKVN / 32);        // 128 row blocks of 32
  int nt = wid % (QKVN / 32);        // 96 col blocks of 32
  const _Float16* arow = xh + (size_t)mt * 32 * H_;
  const _Float16* brow = wT + (size_t)nt * 32 * H_;

  f8 c00 = {}, c01 = {}, c10 = {}, c11 = {};
  for (int k0 = 0; k0 < H_; k0 += 32) {
    h16 a0 = load_frag(arow, H_, k0, lane);
    h16 a1 = load_frag(arow + (size_t)16 * H_, H_, k0, lane);
    h16 b0 = load_frag(brow, H_, k0, lane);
    h16 b1 = load_frag(brow + (size_t)16 * H_, H_, k0, lane);
    if (k0 + 32 < H_) {
      __builtin_prefetch(arow + (size_t)(lane & 15) * H_ + k0 + 32, 0, 3);
      __builtin_prefetch(brow + (size_t)(lane & 15) * H_ + k0 + 32, 0, 3);
    }
    c00 = wmma_f16(a0, b0, c00);
    c01 = wmma_f16(a0, b1, c01);
    c10 = wmma_f16(a1, b0, c10);
    c11 = wmma_f16(a1, b1, c11);
  }

#pragma unroll
  for (int j = 0; j < 2; ++j) {
    int n     = nt * 32 + j * 16 + (lane & 15);
    int which = n >> 10;             // 0=q 1=k 2=v (uniform per subtile)
    int rem   = n & (H_ - 1);
    int head  = rem >> 6;            // uniform per subtile
    int d     = rem & (HD_ - 1);
    rope_store(j ? c01 : c00, mt * 32 + 0,  lane, which, head, d, rcos, rsin, qb, kb, vtb);
    rope_store(j ? c11 : c10, mt * 32 + 16, lane, which, head, d, rcos, rsin, qb, kb, vtb);
  }
}

// ---------------------------------------------------------------------------
// Flash attention: one wave handles 16 queries of one head, looping over all
// 2048 keys in 32-key chunks with online softmax. 8 WMMAs per chunk; Q frags
// stay in registers for the whole pass.
// ---------------------------------------------------------------------------
__global__ void k_attn(const _Float16* __restrict__ qb,
                       const _Float16* __restrict__ kb,
                       const _Float16* __restrict__ vtb,
                       _Float16* __restrict__ att) {          // [4096 x 1024]
  __shared__ _Float16 plds[4][16 * 32];   // per-wave P tile (C-layout -> A-frag)
  int lane  = threadIdx.x & 31;
  int wslot = threadIdx.x >> 5;
  int wid   = blockIdx.x * (blockDim.x >> 5) + wslot;
  int head  = wid >> 7;              // 0..31  (B*NH)
  int qt    = wid & 127;             // query tile within head
  const _Float16* qh = qb  + ((size_t)head * T_ + qt * 16) * HD_;
  const _Float16* kh = kb  + (size_t)head * T_ * HD_;
  const _Float16* vt = vtb + (size_t)head * HD_ * T_;
  _Float16* pl = plds[wslot];
  int hi = lane >> 4;

  h16 qa0 = load_frag(qh, HD_, 0, lane);
  h16 qa1 = load_frag(qh, HD_, 32, lane);
  f8 o0 = {}, o1 = {}, o2 = {}, o3 = {};
  float m[8], lsum[8];
#pragma unroll
  for (int r = 0; r < 8; ++r) { m[r] = -3.0e38f; lsum[r] = 0.f; }

  for (int kc = 0; kc < T_; kc += 32) {
    // scores for keys [kc, kc+32): two 16-key tiles, K=HD split in 2 WMMAs
    f8 s0 = {}, s1 = {};
    s0 = wmma_f16(qa0, load_frag(kh + (size_t)kc * HD_, HD_, 0, lane), s0);
    s0 = wmma_f16(qa1, load_frag(kh + (size_t)kc * HD_, HD_, 32, lane), s0);
    s1 = wmma_f16(qa0, load_frag(kh + (size_t)(kc + 16) * HD_, HD_, 0, lane), s1);
    s1 = wmma_f16(qa1, load_frag(kh + (size_t)(kc + 16) * HD_, HD_, 32, lane), s1);

    float cm[8], corr[8], rsum[8];
#pragma unroll
    for (int r = 0; r < 8; ++r) cm[r] = fmaxf(s0[r], s1[r]);
#pragma unroll
    for (int off = 1; off < 16; off <<= 1) {
#pragma unroll
      for (int r = 0; r < 8; ++r) cm[r] = fmaxf(cm[r], __shfl_xor(cm[r], off, 32));
    }
#pragma unroll
    for (int r = 0; r < 8; ++r) {
      float mn = fmaxf(m[r], cm[r]);
      corr[r]  = __expf(m[r] - mn);
      m[r]     = mn;
      s0[r]    = __expf(s0[r] - mn);
      s1[r]    = __expf(s1[r] - mn);
      rsum[r]  = s0[r] + s1[r];
    }
#pragma unroll
    for (int off = 1; off < 16; off <<= 1) {
#pragma unroll
      for (int r = 0; r < 8; ++r) rsum[r] += __shfl_xor(rsum[r], off, 32);
    }
#pragma unroll
    for (int r = 0; r < 8; ++r) {
      lsum[r] = lsum[r] * corr[r] + rsum[r];
      o0[r] *= corr[r]; o1[r] *= corr[r]; o2[r] *= corr[r]; o3[r] *= corr[r];
      // stage P (f16) into LDS: row = query M, col = key index within chunk
      pl[(r + 8 * hi) * 32 + (lane & 15)]      = (_Float16)s0[r];
      pl[(r + 8 * hi) * 32 + 16 + (lane & 15)] = (_Float16)s1[r];
    }
    asm volatile("s_wait_dscnt 0" ::: "memory");
    h16 pa = load_frag(pl, 32, 0, lane);     // P as 16x32 A-fragment
    // O += P @ V   (B-frag from transposed V: ld = T_)
    o0 = wmma_f16(pa, load_frag(vt + (size_t)0 * 16 * T_ + kc, T_, 0, lane), o0);
    o1 = wmma_f16(pa, load_frag(vt + (size_t)1 * 16 * T_ + kc, T_, 0, lane), o1);
    o2 = wmma_f16(pa, load_frag(vt + (size_t)2 * 16 * T_ + kc, T_, 0, lane), o2);
    o3 = wmma_f16(pa, load_frag(vt + (size_t)3 * 16 * T_ + kc, T_, 0, lane), o3);
  }

  int b_ = head >> 4;
  int hl = head & (NH_ - 1);
#pragma unroll
  for (int r = 0; r < 8; ++r) {
    float inv = 1.0f / lsum[r];
    size_t row  = (size_t)b_ * T_ + qt * 16 + r + 8 * hi;
    size_t colb = (size_t)hl * HD_ + (lane & 15);
    att[row * H_ + colb + 0]  = (_Float16)(o0[r] * inv);
    att[row * H_ + colb + 16] = (_Float16)(o1[r] * inv);
    att[row * H_ + colb + 32] = (_Float16)(o2[r] * inv);
    att[row * H_ + colb + 48] = (_Float16)(o3[r] * inv);
  }
}

// ---------------------------------------------------------------------------
// Output projection: out = att @ w_proj + b_proj (fp32 result).
// 32x32 register block per wave, same scheme as the QKV GEMM.
// ---------------------------------------------------------------------------
__global__ void k_proj(const _Float16* __restrict__ att,
                       const _Float16* __restrict__ wT,   // [1024 x 1024] transposed
                       const float* __restrict__ bias,
                       float* __restrict__ out) {
  int lane = threadIdx.x & 31;
  int wid  = blockIdx.x * (blockDim.x >> 5) + (threadIdx.x >> 5);
  int mt = wid >> 5;     // 128 row blocks of 32
  int nt = wid & 31;     // 32 col blocks of 32
  const _Float16* arow = att + (size_t)mt * 32 * H_;
  const _Float16* brow = wT  + (size_t)nt * 32 * H_;
  f8 c00 = {}, c01 = {}, c10 = {}, c11 = {};
  for (int k0 = 0; k0 < H_; k0 += 32) {
    h16 a0 = load_frag(arow, H_, k0, lane);
    h16 a1 = load_frag(arow + (size_t)16 * H_, H_, k0, lane);
    h16 b0 = load_frag(brow, H_, k0, lane);
    h16 b1 = load_frag(brow + (size_t)16 * H_, H_, k0, lane);
    if (k0 + 32 < H_) {
      __builtin_prefetch(arow + (size_t)(lane & 15) * H_ + k0 + 32, 0, 3);
      __builtin_prefetch(brow + (size_t)(lane & 15) * H_ + k0 + 32, 0, 3);
    }
    c00 = wmma_f16(a0, b0, c00);
    c01 = wmma_f16(a0, b1, c01);
    c10 = wmma_f16(a1, b0, c10);
    c11 = wmma_f16(a1, b1, c11);
  }
  int hi = lane >> 4;
#pragma unroll
  for (int j = 0; j < 2; ++j) {
    int col  = nt * 32 + j * 16 + (lane & 15);
    float bv = bias[col];
#pragma unroll
    for (int i = 0; i < 2; ++i) {
      f8 acc = j ? (i ? c11 : c01) : (i ? c10 : c00);
#pragma unroll
      for (int r = 0; r < 8; ++r) {
        int row = mt * 32 + i * 16 + r + 8 * hi;
        out[(size_t)row * H_ + col] = acc[r] + bv;
      }
    }
  }
}

// ---------------------------------------------------------------------------
extern "C" void kernel_launch(void* const* d_in, const int* in_sizes, int n_in,
                              void* d_out, int out_size, void* d_ws, size_t ws_size,
                              hipStream_t stream) {
  const float* x    = (const float*)d_in[0];
  const float* wqkv = (const float*)d_in[1];
  const float* wprj = (const float*)d_in[2];
  const float* bprj = (const float*)d_in[3];
  const float* rc   = (const float*)d_in[4];
  const float* rs   = (const float*)d_in[5];

  _Float16* ws  = (_Float16*)d_ws;
  _Float16* xh  = ws + OFF_XH;
  _Float16* wqT = ws + OFF_WQKV;
  _Float16* wpT = ws + OFF_WPRJ;
  _Float16* qb  = ws + OFF_Q;
  _Float16* kb  = ws + OFF_K;
  _Float16* vtb = ws + OFF_VT;
  _Float16* att = ws + OFF_ATT;

  {
    int n = ROWS * H_;
    k_cast<<<(n + 255) / 256, 256, 0, stream>>>(x, xh, n);
  }
  {
    int n = H_ * QKVN;
    k_transpose<<<(n + 255) / 256, 256, 0, stream>>>(wqkv, wqT, H_, QKVN);
  }
  {
    int n = H_ * H_;
    k_transpose<<<(n + 255) / 256, 256, 0, stream>>>(wprj, wpT, H_, H_);
  }
  {
    int tiles = (ROWS / 32) * (QKVN / 32);   // 12288 wave-tiles (32x32 each)
    k_qkv_rope<<<tiles / 4, 128, 0, stream>>>(xh, wqT, rc, rs, qb, kb, vtb);
  }
  {
    int tiles = B_ * NH_ * (T_ / 16);        // 4096 wave-tiles
    k_attn<<<tiles / 4, 128, 0, stream>>>(qb, kb, vtb, att);
  }
  {
    int tiles = (ROWS / 32) * (H_ / 32);     // 4096 wave-tiles (32x32 each)
    k_proj<<<tiles / 4, 128, 0, stream>>>(att, wpT, bprj, (float*)d_out);
  }
}
